// SOMNetwork_34565896799009
// MI455X (gfx1250) — matrix operand order repaired and checked by reading, not compiled, not run on hardware
//
#include <hip/hip_runtime.h>
#include <hip/hip_bf16.h>

// ---------------- problem constants ----------------
#define Bsz 16
#define Cin 3
#define Hin 224
#define Win 224
#define Oout 128
#define KH 7
#define KW 7
#define Ktap 49            // KH*KW
#define Kpad 64            // padded K for WMMA (2 chunks of 32)
#define OH 218
#define OW 218
#define MT 16              // M-tile (pixels per block)
#define MT_PER_ROW 14      // ceil(218/16)

// workspace layout (bytes)
#define GRAY_BYTES (Bsz * Hin * Win * 4)          // 3,211,264
#define WPACK_BYTES (Oout * Kpad * 2)             // 16,384
#define WSQ_OFF (GRAY_BYTES + WPACK_BYTES)        // wsq[0..127] + inv2s2 at [128]

typedef __attribute__((ext_vector_type(16))) _Float16 v16h;
typedef __attribute__((ext_vector_type(8)))  float    v8f;

// ---------------- prep: grayscale (ITU-R 601) ----------------
__global__ void som_gray_kernel(const float* __restrict__ x, float* __restrict__ gray) {
    int idx = blockIdx.x * blockDim.x + threadIdx.x;
    const int HW = Hin * Win;
    if (idx >= Bsz * HW) return;
    int b = idx / HW, hw = idx - b * HW;
    const float* xb = x + (size_t)b * Cin * HW + hw;
    float g = 0.299f * xb[0] + 0.587f * xb[HW] + 0.114f * xb[2 * HW];
    gray[idx] = g;
}

// ---------------- prep: pack -2*w into WMMA B-lane layout (f16, K padded to 64)
// B fragment layout (16-bit, 32x16): lane l holds column n = l&15,
// lane<16: K = 0..7 then 16..23; lane>=16: K = 8..15 then 24..31.
__global__ void som_packw_kernel(const float* __restrict__ w, _Float16* __restrict__ wpack) {
    int t = blockIdx.x * blockDim.x + threadIdx.x;
    if (t >= Oout * Kpad) return;
    int n = t >> 6;          // filter
    int k = t & 63;          // padded K index
    float v = (k < Ktap) ? (-2.0f * w[n * Ktap + k]) : 0.0f;
    int nt = n >> 4;         // N-tile (wave) 0..7
    int c  = k >> 5;         // K-chunk 0..1
    int kin = k & 31;
    int lane = (n & 15) | ((kin & 8) << 1);
    int h = (kin & 7) | ((kin >> 1) & 8);
    wpack[(((nt * 2 + c) * 32) + lane) * 16 + h] = (_Float16)v;
}

// ---------------- prep: ||w||^2 per filter (f32) + hoisted 1/(2*std^2) ----------------
__global__ void som_wsq_kernel(const float* __restrict__ w, const float* __restrict__ stdp,
                               float* __restrict__ wsq) {
    int n = threadIdx.x;
    if (n >= Oout) return;
    float s = 0.f;
    #pragma unroll
    for (int k = 0; k < Ktap; ++k) { float v = w[n * Ktap + k]; s += v * v; }
    wsq[n] = s;
    if (n == 0) {
        float sd = stdp[0];
        wsq[Oout] = 1.0f / (2.0f * sd * sd);   // divide ONCE here, not per main-thread
    }
}

// ---------------- main: WMMA distance + RBF ----------------
// grid.x = Bsz * OH * MT_PER_ROW, block = 256 (8 wave32)
// block -> 16 consecutive output pixels (one row segment) x all 128 filters
__global__ void __launch_bounds__(256)
som_wmma_kernel(const float* __restrict__ gray,
                const _Float16* __restrict__ wpack,
                const float* __restrict__ wsq,
                float* __restrict__ out) {
    __shared__ __align__(16) _Float16 A_lds[MT * Kpad];  // 2 KB
    __shared__ float psq[MT];

    int bx  = blockIdx.x;
    int mt  = bx % MT_PER_ROW;
    int tmp = bx / MT_PER_ROW;
    int oh  = tmp % OH;
    int b   = tmp / OH;
    int ow0 = mt * MT;

    int t = threadIdx.x;

    // ---- stage A tile (16 pixels x 64 K) into LDS, accumulate ||p||^2 in f32
    if (t < MT) psq[t] = 0.f;
    __syncthreads();

    int m  = t & 15;       // pixel within tile
    int kq = t >> 4;       // 0..15, each covers 4 K slots
    bool pvalid = (ow0 + m) < OW;
    float ss = 0.f;
    #pragma unroll
    for (int i = 0; i < 4; ++i) {
        int k = kq * 4 + i;
        float v = 0.f;
        if (k < Ktap && pvalid) {
            int kh = k / KW, kw = k - kh * KW;
            v = gray[((b * Hin) + (oh + kh)) * Win + (ow0 + m + kw)];
        }
        A_lds[m * Kpad + k] = (_Float16)v;
        ss += v * v;
    }
    atomicAdd(&psq[m], ss);   // ds_add_f32
    __syncthreads();

    // ---- per-wave fragments
    int lane = t & 31;
    int nt   = t >> 5;       // N-tile 0..7 (filters nt*16 .. nt*16+15)

    // A fragments: ISA 16-bit A 16x32 layout
    // lane<16: row=lane,    halves K = kb+0..7 then kb+16..23
    // lane>=16: row=lane-16, halves K = kb+8..15 then kb+24..31
    int sel  = (lane < 16) ? 0 : 8;
    int mrow = lane & 15;
    const uint4* arow = (const uint4*)(&A_lds[mrow * Kpad]);  // 8 halves per uint4
    union AF { uint4 q[2]; v16h h; } a0, a1;
    a0.q[0] = arow[(0  + sel) >> 3];
    a0.q[1] = arow[(16 + sel) >> 3];
    a1.q[0] = arow[(32 + sel) >> 3];
    a1.q[1] = arow[(48 + sel) >> 3];

    // B fragments: pre-packed, 32 contiguous bytes per lane per chunk
    const uint4* wb = (const uint4*)wpack;
    union BF { uint4 q[2]; v16h h; } b0, b1;
    int base0 = (((nt * 2 + 0) * 32) + lane) * 2;
    int base1 = (((nt * 2 + 1) * 32) + lane) * 2;
    b0.q[0] = wb[base0 + 0]; b0.q[1] = wb[base0 + 1];
    b1.q[0] = wb[base1 + 0]; b1.q[1] = wb[base1 + 1];

    // ---- D = A x B (+C): acc = -2 * (p . w)
    v8f acc = {};
    acc = __builtin_amdgcn_wmma_f32_16x16x32_f16(false, a0.h, false, b0.h,
                                                 (short)0, acc, false, false);
    acc = __builtin_amdgcn_wmma_f32_16x16x32_f16(false, a1.h, false, b1.h,
                                                 (short)0, acc, false, false);

    // ---- epilogue: d2 = acc + ||p||^2 + ||w||^2 ; out = exp(-d2 * inv2s2)
    float inv2s2 = wsq[Oout];           // precomputed 1/(2*std^2), scalar load

    int n  = nt * 16 + (lane & 15);
    int m0 = (lane < 16) ? 0 : 8;       // C/D layout: VGPR j -> M = j + m0
    float wsq_n = wsq[n];
    size_t base = (((size_t)b * Oout + n) * OH + oh) * OW + ow0;

    float r[8];
    #pragma unroll
    for (int j = 0; j < 8; ++j) {
        float d2 = acc[j] + psq[m0 + j] + wsq_n;
        d2 = fmaxf(d2, 0.0f);
        r[j] = __expf(-d2 * inv2s2);
    }

    if (ow0 + MT <= OW) {
        // full tile: 8 consecutive floats per lane, 8B-aligned -> float2 stores
        float2* p2 = (float2*)(out + base + m0);
        p2[0] = make_float2(r[0], r[1]);
        p2[1] = make_float2(r[2], r[3]);
        p2[2] = make_float2(r[4], r[5]);
        p2[3] = make_float2(r[6], r[7]);
    } else {
        #pragma unroll
        for (int j = 0; j < 8; ++j) {
            int mm = m0 + j;
            if (ow0 + mm < OW) out[base + mm] = r[j];
        }
    }
}

// ---------------- launch ----------------
extern "C" void kernel_launch(void* const* d_in, const int* in_sizes, int n_in,
                              void* d_out, int out_size, void* d_ws, size_t ws_size,
                              hipStream_t stream) {
    const float* x    = (const float*)d_in[0];   // [16,3,224,224]
    const float* w    = (const float*)d_in[1];   // [128,1,7,7]
    const float* stdp = (const float*)d_in[2];   // [1]
    float* out = (float*)d_out;                  // [16,128,218,218]

    char* ws = (char*)d_ws;
    float*    gray  = (float*)ws;
    _Float16* wpack = (_Float16*)(ws + GRAY_BYTES);
    float*    wsq   = (float*)(ws + WSQ_OFF);    // 128 norms + inv2s2

    // prep
    {
        int n = Bsz * Hin * Win;
        som_gray_kernel<<<(n + 255) / 256, 256, 0, stream>>>(x, gray);
    }
    {
        int n = Oout * Kpad;
        som_packw_kernel<<<(n + 255) / 256, 256, 0, stream>>>(w, wpack);
    }
    som_wsq_kernel<<<1, 128, 0, stream>>>(w, stdp, wsq);

    // main
    dim3 grid(Bsz * OH * MT_PER_ROW);   // 48,832 blocks
    som_wmma_kernel<<<grid, 256, 0, stream>>>(gray, wpack, wsq, out);
}
